// TransformerEncoderLayer_22677427323305
// MI455X (gfx1250) — matrix operand order repaired and checked by reading, not compiled, not run on hardware
//
#include <hip/hip_runtime.h>
#include <hip/hip_bf16.h>

// ---------------------------------------------------------------------------
// Types for CDNA5 WMMA (wave32): D(16x16 f32) = A(16x32 bf16) x B(32x16 bf16) + C
// ---------------------------------------------------------------------------
typedef __attribute__((ext_vector_type(16))) __bf16 v16bf;
typedef __attribute__((ext_vector_type(8)))  __bf16 v8bf;
typedef __attribute__((ext_vector_type(8)))  float  v8f;

union ABFrag { v16bf v; v8bf h[2]; };

#define B_BATCH 2
#define T_SEQ   2048
#define D_MODEL 1024
#define NHEAD   16
#define HEAD_DIM 64
#define INT_DIM 4096
#define MT      (B_BATCH * T_SEQ)   // 4096 rows

static __device__ __forceinline__ v8f wmma_bf16(const v16bf& a, const v16bf& b, v8f c) {
    return __builtin_amdgcn_wmma_f32_16x16x32_bf16(false, a, false, b, (short)0, c, false, false);
}

// ---------------------------------------------------------------------------
// fp32 -> bf16 elementwise convert
// ---------------------------------------------------------------------------
__global__ void k_f32_to_bf16(const float* __restrict__ in, __bf16* __restrict__ out, int n) {
    int i = blockIdx.x * blockDim.x + threadIdx.x;
    int stride = gridDim.x * blockDim.x;
    for (; i < n; i += stride) out[i] = (__bf16)in[i];
}

// ---------------------------------------------------------------------------
// Tiled BF16 WMMA GEMM:  C[M,N] = A[M,K] @ B[K,N] + bias  (+residual / relu)
// Compile-time specialized (RESID / RELU / OUT_BF16) -> branch-free epilogue.
// Block tile 128x128, BK=32, 256 threads = 8 waves (4 along M x 2 along N),
// wave tile 32x64 = 2x4 WMMA fragments.
// Double-buffered LDS (2 x 16KB): prefetch tile i+1 into registers while
// WMMAs consume tile i; one barrier per K-step. Tile i+2 software-prefetched
// into GL2 via global_prefetch.
// ---------------------------------------------------------------------------
template<bool RESID, bool RELU, bool OUT_BF16>
__global__ __launch_bounds__(256)
void k_gemm(const __bf16* __restrict__ A, const __bf16* __restrict__ Bm,
            const float* __restrict__ bias, const float* __restrict__ resid,
            float* __restrict__ outF, __bf16* __restrict__ outB,
            int M, int N, int K) {
    __shared__ __bf16 As[2][128 * 32];   // [buf][m][k]
    __shared__ __bf16 Bs[2][128 * 32];   // [buf][n][k]  (transposed tile)

    const int tid  = threadIdx.x;
    const int wave = tid >> 5;
    const int lane = tid & 31;
    const int half = lane >> 4;
    const int l15  = lane & 15;
    const int waveM = wave & 3;       // 0..3  -> 32-row slab
    const int waveN = wave >> 2;      // 0..1  -> 64-col slab

    const int m0 = blockIdx.y * 128;
    const int n0 = blockIdx.x * 128;

    // per-thread staging coordinates
    const int ar = tid >> 1, ac = (tid & 1) * 16;          // A: row, col
    const int bk = tid >> 3, bn = (tid & 7) * 16;          // B: k-row, n-col

    v8f acc[2][4] = {};
    v8bf ra[2], rb[2];

    auto loadG = [&](int k0) {
        const v8bf* sa = (const v8bf*)(A + (size_t)(m0 + ar) * K + k0 + ac);
        ra[0] = sa[0]; ra[1] = sa[1];
        const v8bf* sb = (const v8bf*)(Bm + (size_t)(k0 + bk) * N + n0 + bn);
        rb[0] = sb[0]; rb[1] = sb[1];
    };
    auto storeL = [&](int buf) {
        *(v8bf*)&As[buf][ar * 32 + ac]     = ra[0];
        *(v8bf*)&As[buf][ar * 32 + ac + 8] = ra[1];
#pragma unroll
        for (int i = 0; i < 8; i++) {
            Bs[buf][(bn + i) * 32 + bk]     = rb[0][i];
            Bs[buf][(bn + 8 + i) * 32 + bk] = rb[1][i];
        }
    };

    const int nK = K >> 5;
    loadG(0);
    storeL(0);
    __syncthreads();

    for (int i = 0; i < nK; i++) {
        const int cur = i & 1, nxt = cur ^ 1;
        const bool more = (i + 1) < nK;
        if (more) loadG((i + 1) * 32);                       // overlap with WMMA
        if (i + 2 < nK) {                                    // pull tile i+2 into GL2
            __builtin_prefetch(A + (size_t)(m0 + ar) * K + (i + 2) * 32 + ac, 0, 1);
            __builtin_prefetch(Bm + (size_t)((i + 2) * 32 + bk) * N + n0 + bn, 0, 1);
        }

        // ---- A fragments for this wave (2 row tiles) ----
        ABFrag a[2];
#pragma unroll
        for (int mt = 0; mt < 2; mt++) {
            int row = waveM * 32 + mt * 16 + l15;
            a[mt].h[0] = *(const v8bf*)&As[cur][row * 32 + half * 16];
            a[mt].h[1] = *(const v8bf*)&As[cur][row * 32 + half * 16 + 8];
        }
        // ---- 4 col tiles x 2 row tiles WMMA ----
#pragma unroll
        for (int nt = 0; nt < 4; nt++) {
            int col = waveN * 64 + nt * 16 + l15;
            v16bf b = *(const v16bf*)&Bs[cur][col * 32 + half * 16];
#pragma unroll
            for (int mt = 0; mt < 2; mt++)
                acc[mt][nt] = wmma_bf16(a[mt].v, b, acc[mt][nt]);
        }

        if (more) storeL(nxt);      // disjoint from buf[cur] readers
        __syncthreads();
    }

    // ---- branch-free epilogue ----
#pragma unroll
    for (int mt = 0; mt < 2; mt++) {
#pragma unroll
        for (int nt = 0; nt < 4; nt++) {
            int gm = m0 + waveM * 32 + mt * 16 + half * 8;
            int gn = n0 + waveN * 64 + nt * 16 + l15;
            float bv = bias[gn];
#pragma unroll
            for (int r = 0; r < 8; r++) {
                size_t idx = (size_t)(gm + r) * N + gn;
                float v = acc[mt][nt][r] + bv;
                if constexpr (RESID) v += resid[idx];
                if constexpr (RELU)  v = fmaxf(v, 0.0f);
                if constexpr (OUT_BF16) outB[idx] = (__bf16)v;
                else                    outF[idx] = v;
            }
        }
    }
}

// ---------------------------------------------------------------------------
// Flash attention. Q/K/V in bf16, stored as [B*T, D] views ([b,t,h,dh]).
// Block = 128 threads (4 waves), handles 64 query rows for one (b,h).
// Wave = 16 query rows. Online softmax over key blocks of 64.
// Scores: A = Q frags (regs), B = K rows straight from global (key rows are
// contiguous bf16 == B-operand layout). P@V: P bf16 in LDS, V staged
// transposed ([dh][key]) in LDS.
// ---------------------------------------------------------------------------
__global__ __launch_bounds__(128)
void k_attention(const __bf16* __restrict__ Q, const __bf16* __restrict__ K,
                 const __bf16* __restrict__ V, __bf16* __restrict__ O) {
    __shared__ __bf16 Vt[64 * 64];        // [dh][key]
    __shared__ float  S[4][16 * 64];      // per-wave score tile
    __shared__ __bf16 P[4][16 * 64];      // per-wave prob tile (bf16)
    __shared__ float  rowM[4][16], rowL[4][16], rowA[4][16];

    const int bh   = blockIdx.y;
    const int b    = bh >> 4;
    const int h    = bh & 15;
    const int q0   = blockIdx.x * 64;
    const int wave = threadIdx.x >> 5;
    const int lane = threadIdx.x & 31;
    const int half = lane >> 4;
    const int l15  = lane & 15;

    // Q fragments (A operand), per-lane row = l15
    ABFrag qf[2];
    {
        const __bf16* qp = Q + ((size_t)b * T_SEQ + q0 + wave * 16 + l15) * D_MODEL + h * 64;
#pragma unroll
        for (int kd = 0; kd < 2; kd++) {
            qf[kd].h[0] = *(const v8bf*)(qp + kd * 32 + half * 16);
            qf[kd].h[1] = *(const v8bf*)(qp + kd * 32 + half * 16 + 8);
        }
    }

    v8f oacc[4] = {};
    if (lane < 16) { rowM[wave][lane] = -1e30f; rowL[wave][lane] = 0.0f; }
    const float scale = 0.125f;   // 1/sqrt(64)

    for (int kb = 0; kb < T_SEQ; kb += 64) {
        __syncthreads();  // previous P@V done before Vt is overwritten
        // ---- stage V block transposed: 64 keys x 64 dh ----
#pragma unroll
        for (int i = 0; i < 4; i++) {
            int c   = threadIdx.x + i * 128;    // 0..511 chunks of 8
            int key = c >> 3;
            int dh0 = (c & 7) * 8;
            v8bf vv = *(const v8bf*)(V + ((size_t)b * T_SEQ + kb + key) * D_MODEL + h * 64 + dh0);
#pragma unroll
            for (int j = 0; j < 8; j++) Vt[(dh0 + j) * 64 + key] = vv[j];
        }
        __syncthreads();

        // ---- scores: 16x64 tile per wave ----
#pragma unroll
        for (int nt = 0; nt < 4; nt++) {
            v8f c = {};
            const __bf16* kp = K + ((size_t)b * T_SEQ + kb + nt * 16 + l15) * D_MODEL + h * 64;
#pragma unroll
            for (int kd = 0; kd < 2; kd++) {
                v16bf bfrag = *(const v16bf*)(kp + kd * 32 + half * 16);
                c = wmma_bf16(qf[kd].v, bfrag, c);
            }
#pragma unroll
            for (int r = 0; r < 8; r++)
                S[wave][(r + half * 8) * 64 + nt * 16 + l15] = c[r] * scale;
        }
        asm volatile("s_wait_dscnt 0" ::: "memory");

        // ---- online softmax: lanes 0..15 each own one row ----
        if (lane < 16) {
            float* srow = &S[wave][lane * 64];
            float m_old = rowM[wave][lane];
            float bm = -1e30f;
#pragma unroll 8
            for (int j = 0; j < 64; j++) bm = fmaxf(bm, srow[j]);
            float mnew  = fmaxf(m_old, bm);
            float alpha = __expf(m_old - mnew);
            float sum = 0.0f;
#pragma unroll 8
            for (int j = 0; j < 64; j++) {
                float p = __expf(srow[j] - mnew);
                sum += p;
                P[wave][lane * 64 + j] = (__bf16)p;
            }
            rowL[wave][lane] = rowL[wave][lane] * alpha + sum;
            rowM[wave][lane] = mnew;
            rowA[wave][lane] = alpha;
        }
        asm volatile("s_wait_dscnt 0" ::: "memory");

        // ---- rescale O accumulators per row ----
        float al[8];
#pragma unroll
        for (int r = 0; r < 8; r++) al[r] = rowA[wave][r + half * 8];
#pragma unroll
        for (int nt = 0; nt < 4; nt++)
#pragma unroll
            for (int r = 0; r < 8; r++) oacc[nt][r] *= al[r];

        // ---- P @ V ----
        ABFrag pf[2];
#pragma unroll
        for (int kd = 0; kd < 2; kd++) {
            pf[kd].h[0] = *(const v8bf*)&P[wave][l15 * 64 + kd * 32 + half * 16];
            pf[kd].h[1] = *(const v8bf*)&P[wave][l15 * 64 + kd * 32 + half * 16 + 8];
        }
#pragma unroll
        for (int nt = 0; nt < 4; nt++) {
            int dh = nt * 16 + l15;
#pragma unroll
            for (int kd = 0; kd < 2; kd++) {
                v16bf vfrag = *(const v16bf*)&Vt[dh * 64 + kd * 32 + half * 16];
                oacc[nt] = wmma_bf16(pf[kd].v, vfrag, oacc[nt]);
            }
        }
    }

    // ---- epilogue: O /= l, store bf16 [b,t,h,dh] ----
    float li[8];
#pragma unroll
    for (int r = 0; r < 8; r++) li[r] = rowL[wave][r + half * 8];
#pragma unroll
    for (int nt = 0; nt < 4; nt++)
#pragma unroll
        for (int r = 0; r < 8; r++) {
            int row = q0 + wave * 16 + r + half * 8;
            O[((size_t)b * T_SEQ + row) * D_MODEL + h * 64 + nt * 16 + l15] =
                (__bf16)(oacc[nt][r] / li[r]);
        }
}

// ---------------------------------------------------------------------------
// Row LayerNorm over D=1024 (one block per row, 256 threads x 4 elems),
// optionally emitting a bf16 copy for the next GEMM.
// ---------------------------------------------------------------------------
template<bool EMIT_BF16>
__global__ __launch_bounds__(256)
void k_layernorm(const float* __restrict__ in, const float* __restrict__ g,
                 const float* __restrict__ beta, float* __restrict__ outF,
                 __bf16* __restrict__ outB) {
    __shared__ float sA[256], sB[256];
    __shared__ float shMean, shRstd;
    const int row = blockIdx.x;
    const int tid = threadIdx.x;
    const float* p = in + (size_t)row * D_MODEL;

    float s = 0.0f, s2 = 0.0f;
#pragma unroll
    for (int i = 0; i < 4; i++) {
        float v = p[tid + i * 256];
        s += v; s2 += v * v;
    }
    sA[tid] = s; sB[tid] = s2;
    __syncthreads();
    for (int off = 128; off > 0; off >>= 1) {
        if (tid < off) { sA[tid] += sA[tid + off]; sB[tid] += sB[tid + off]; }
        __syncthreads();
    }
    if (tid == 0) {
        float mean = sA[0] * (1.0f / D_MODEL);
        float var  = sB[0] * (1.0f / D_MODEL) - mean * mean;
        shMean = mean;
        shRstd = rsqrtf(var + 1e-5f);
    }
    __syncthreads();
    float mean = shMean, rstd = shRstd;
#pragma unroll
    for (int i = 0; i < 4; i++) {
        int c = tid + i * 256;
        float v = (p[c] - mean) * rstd * g[c] + beta[c];
        outF[(size_t)row * D_MODEL + c] = v;
        if constexpr (EMIT_BF16) outB[(size_t)row * D_MODEL + c] = (__bf16)v;
    }
}

// ---------------------------------------------------------------------------
// Host launcher
// ---------------------------------------------------------------------------
extern "C" void kernel_launch(void* const* d_in, const int* in_sizes, int n_in,
                              void* d_out, int out_size, void* d_ws, size_t ws_size,
                              hipStream_t stream) {
    const float* x     = (const float*)d_in[0];
    const float* Wq    = (const float*)d_in[1];
    const float* bq    = (const float*)d_in[2];
    const float* Wk    = (const float*)d_in[3];
    const float* bk    = (const float*)d_in[4];
    const float* Wv    = (const float*)d_in[5];
    const float* bv    = (const float*)d_in[6];
    const float* Wo    = (const float*)d_in[7];
    const float* bo    = (const float*)d_in[8];
    const float* ln1_g = (const float*)d_in[9];
    const float* ln1_b = (const float*)d_in[10];
    const float* ln2_g = (const float*)d_in[11];
    const float* ln2_b = (const float*)d_in[12];
    const float* W1    = (const float*)d_in[13];
    const float* b1    = (const float*)d_in[14];
    const float* W2    = (const float*)d_in[15];
    const float* b2    = (const float*)d_in[16];
    float* out = (float*)d_out;

    const size_t MB = 1024 * 1024;
    char* ws = (char*)d_ws;
    __bf16* xb    = (__bf16*)(ws + 0);        // 8MB  (reused as attention O)
    __bf16* Wq_b  = (__bf16*)(ws + 8  * MB);  // 2MB
    __bf16* Wk_b  = (__bf16*)(ws + 10 * MB);
    __bf16* Wv_b  = (__bf16*)(ws + 12 * MB);
    __bf16* Wo_b  = (__bf16*)(ws + 14 * MB);
    __bf16* W1_b  = (__bf16*)(ws + 16 * MB);  // 8MB
    __bf16* W2_b  = (__bf16*)(ws + 24 * MB);  // 8MB
    __bf16* Qb    = (__bf16*)(ws + 32 * MB);  // 8MB
    __bf16* Kb    = (__bf16*)(ws + 40 * MB);
    __bf16* Vb    = (__bf16*)(ws + 48 * MB);
    float*  h_pre = (float*) (ws + 56 * MB);  // 16MB (reused as ff_pre)
    float*  hbuf  = (float*) (ws + 72 * MB);  // 16MB
    __bf16* h_bf  = (__bf16*)(ws + 88 * MB);  // 8MB
    __bf16* act   = (__bf16*)(ws + 96 * MB);  // 32MB
    __bf16* Ob    = xb;
    float*  f_pre = h_pre;

    auto conv = [&](const float* src, __bf16* dst, int n) {
        int blocks = (n + 1023) / 1024; if (blocks > 4096) blocks = 4096;
        k_f32_to_bf16<<<blocks, 256, 0, stream>>>(src, dst, n);
    };

    // 1. convert activations + weights to bf16
    conv(x,  xb,   MT * D_MODEL);
    conv(Wq, Wq_b, D_MODEL * D_MODEL);
    conv(Wk, Wk_b, D_MODEL * D_MODEL);
    conv(Wv, Wv_b, D_MODEL * D_MODEL);
    conv(Wo, Wo_b, D_MODEL * D_MODEL);
    conv(W1, W1_b, D_MODEL * INT_DIM);
    conv(W2, W2_b, INT_DIM * D_MODEL);

    dim3 blk(256);
    dim3 gD(D_MODEL / 128, MT / 128);   // (8, 32)
    dim3 gF(INT_DIM / 128, MT / 128);   // (32, 32)

    // 2. Q/K/V projections (bf16 out, [b,t,h,dh] view of [MT,D])
    k_gemm<false, false, true><<<gD, blk, 0, stream>>>(xb, Wq_b, bq, nullptr, nullptr, Qb,
                                                       MT, D_MODEL, D_MODEL);
    k_gemm<false, false, true><<<gD, blk, 0, stream>>>(xb, Wk_b, bk, nullptr, nullptr, Kb,
                                                       MT, D_MODEL, D_MODEL);
    k_gemm<false, false, true><<<gD, blk, 0, stream>>>(xb, Wv_b, bv, nullptr, nullptr, Vb,
                                                       MT, D_MODEL, D_MODEL);

    // 3. flash attention -> Ob (bf16, [MT,D])
    k_attention<<<dim3(T_SEQ / 64, B_BATCH * NHEAD), dim3(128), 0, stream>>>(Qb, Kb, Vb, Ob);

    // 4. output projection + residual(x) -> h_pre (f32)
    k_gemm<true, false, false><<<gD, blk, 0, stream>>>(Ob, Wo_b, bo, x, h_pre, nullptr,
                                                       MT, D_MODEL, D_MODEL);

    // 5. LN1 -> h (f32) + h_bf (bf16)
    k_layernorm<true><<<MT, 256, 0, stream>>>(h_pre, ln1_g, ln1_b, hbuf, h_bf);

    // 6. FFN1: relu(h @ W1 + b1) -> act (bf16)
    k_gemm<false, true, true><<<gF, blk, 0, stream>>>(h_bf, W1_b, b1, nullptr, nullptr, act,
                                                      MT, INT_DIM, D_MODEL);

    // 7. FFN2: act @ W2 + b2 + h -> f_pre (f32)
    k_gemm<true, false, false><<<gD, blk, 0, stream>>>(act, W2_b, b2, hbuf, f_pre, nullptr,
                                                       MT, D_MODEL, INT_DIM);

    // 8. LN2 -> d_out (f32)
    k_layernorm<false><<<MT, 256, 0, stream>>>(f_pre, ln2_g, ln2_b, out, nullptr);
}
